// MPNNEncoderVAE_39779987095906
// MI455X (gfx1250) — matrix operand order repaired and checked by reading, not compiled, not run on hardware
//
#include <hip/hip_runtime.h>
#include <hip/hip_bf16.h>

// ---------------------------------------------------------------------------
// Problem constants (match reference)
// ---------------------------------------------------------------------------
constexpr int N_NODES = 10000;
constexpr int E_EDGES = 320000;
constexpr int E2      = E_EDGES + N_NODES;   // with self loops = 330000
constexpr int F_IN    = 128;
constexpr int HID     = 256;
constexpr int HEADS   = 8;
constexpr int CH      = HID / HEADS;         // 32
constexpr int NLAYERS = 3;
constexpr int LAT     = 64;
constexpr int LORA_R  = 8;
constexpr int NGRAPH  = 64;
constexpr float SCALING   = 16.0f / (float)LORA_R;   // 2.0
constexpr float NEG_SLOPE = 0.2f;

typedef __attribute__((ext_vector_type(16))) _Float16 v16h;
typedef __attribute__((ext_vector_type(8)))  float    v8f;

// ---------------------------------------------------------------------------
// One-time fp32 -> f16 weight conversion (done once per launch; removes all
// B-operand cvts and halves B-load bytes in the GEMM hot loop).
// ---------------------------------------------------------------------------
__global__ void cvt_f16(const float* __restrict__ src, _Float16* __restrict__ dst,
                        int n) {
  int i = blockIdx.x * blockDim.x + threadIdx.x;
  if (i < n) dst[i] = (_Float16)src[i];
}

// ---------------------------------------------------------------------------
// WMMA GEMM:  out[n][m] = sum_k in[n][k] * W16[m][k]  (+ bias[m])
// fp32 activations in memory, f16 matrix-core inputs, fp32 accumulate.
//
// Block = (32, 8) = 8 waves. grid.x = nrows/16. Each block computes a
// 16 x 256 output slab: the 16-row A tile is staged ONCE into LDS (converted
// to f16 during staging), then each wave computes 2 adjacent 16x16 m-tiles,
// reusing one A fragment for two WMMAs (register blocking).
//
// LDS row stride = K + 8 halfs: consecutive rows land 4 banks apart, so the
// 16 lanes doing ds_load_b128 of a row cover disjoint 4-bank groups ->
// conflict-free across all 64 banks.
//
// A-fragment layout (16-bit A 16x32): lane half h holds
//   K = kk + 16h + {0..15}   (row = lane%16)
// B-fragment layout (32x16): lane half h holds K = kk + 16h + {0..15}
//   (col = lane%16) -> one aligned 32B v16h load (2x global_load_b128).
// ---------------------------------------------------------------------------
template <int K, bool BIAS>
__global__ void gemm_wmma(const float* __restrict__ in, const _Float16* __restrict__ W16,
                          const float* __restrict__ bias, float* __restrict__ out,
                          int M) {
  constexpr int LDA = K + 8;                 // padded f16 row stride
  __shared__ _Float16 As[16 * LDA];

  const int tid    = threadIdx.y * 32 + threadIdx.x;
  const int tn     = blockIdx.x;             // 16-row slab index
  const int lane16 = threadIdx.x & 15;
  const int half   = threadIdx.x >> 4;       // 0/1

  // ---- cooperative stage: 16 x K fp32 -> f16 in LDS (convert once/block) ----
  for (int idx4 = tid; idx4 < (16 * K) / 4; idx4 += 256) {
    const int r = (idx4 * 4) / K;
    const int k = (idx4 * 4) % K;
    const float4 v = *(const float4*)(in + (size_t)(tn * 16 + r) * K + k);
    _Float16* dp = &As[r * LDA + k];
    dp[0] = (_Float16)v.x;
    dp[1] = (_Float16)v.y;
    dp[2] = (_Float16)v.z;
    dp[3] = (_Float16)v.w;
  }
  __syncthreads();

  // ---- per-wave: two adjacent 16x16 output tiles ----
  const int col0 = (threadIdx.y * 2) * 16 + lane16;
  const int col1 = col0 + 16;
  const _Float16* __restrict__ b0 = W16 + (size_t)col0 * K;
  const _Float16* __restrict__ b1 = W16 + (size_t)col1 * K;
  const _Float16* __restrict__ arow = &As[lane16 * LDA];

  v8f acc0 = {}, acc1 = {};
#pragma unroll
  for (int kk = 0; kk < K; kk += 32) {
    const int ka = kk + half * 16;
    const int kb = kk + half * 16;
    v16h a;
#pragma unroll
    for (int i = 0; i < 8; ++i) {            // 2x ds_load_b128
      a[i]     = arow[ka + i];
      a[i + 8] = arow[ka + 8 + i];
    }
    const v16h bb0 = *(const v16h*)(b0 + kb);  // 2x global_load_b128, no cvt
    const v16h bb1 = *(const v16h*)(b1 + kb);
    acc0 = __builtin_amdgcn_wmma_f32_16x16x32_f16(false, a, false, bb0,
                                                  (short)0, acc0, false, false);
    acc1 = __builtin_amdgcn_wmma_f32_16x16x32_f16(false, a, false, bb1,
                                                  (short)0, acc1, false, false);
  }

  // C/D layout: VGPR r -> row (tn*16 + 8*half + r), col = lane16 column
#pragma unroll
  for (int r = 0; r < 8; ++r) {
    const int row_out = tn * 16 + half * 8 + r;
    float v0 = acc0[r], v1 = acc1[r];
    if (BIAS) { v0 += bias[col0]; v1 += bias[col1]; }
    out[(size_t)row_out * M + col0] = v0;
    out[(size_t)row_out * M + col1] = v1;
  }
}

// ---------------------------------------------------------------------------
// Per-node, per-head attention coefficients: a_s[n,h], a_d[n,h]
// ---------------------------------------------------------------------------
__global__ void att_coef(const float* __restrict__ hw,
                         const float* __restrict__ att_s,
                         const float* __restrict__ att_d,
                         float* __restrict__ as, float* __restrict__ ad) {
  int i = blockIdx.x * blockDim.x + threadIdx.x;   // n*HEADS + h
  if (i >= N_NODES * HEADS) return;
  const int h = i & (HEADS - 1);
  const float* v = hw + (size_t)(i >> 3) * HID + h * CH;
  float s = 0.f, d = 0.f;
#pragma unroll
  for (int c = 0; c < CH; ++c) {
    s += v[c] * att_s[h * CH + c];
    d += v[c] * att_d[h * CH + c];
  }
  as[i] = s;
  ad[i] = d;
}

// ---------------------------------------------------------------------------
// Per-layer init: zero aggregation target, m = -inf, denom = 0
// ---------------------------------------------------------------------------
__global__ void layer_init(float* __restrict__ agg, float* __restrict__ mmax,
                           float* __restrict__ denom) {
  int i = blockIdx.x * blockDim.x + threadIdx.x;
  if (i < N_NODES * HID) agg[i] = 0.f;
  if (i < N_NODES * HEADS) {
    mmax[i]  = -__builtin_inff();
    denom[i] = 0.f;
  }
}

__device__ __forceinline__ void atomicMaxF(float* addr, float v) {
  if (v >= 0.f)
    atomicMax((int*)addr, __float_as_int(v));
  else
    atomicMin((unsigned int*)addr, __float_as_uint(v));
}

// ---------------------------------------------------------------------------
// Edge logits: e = leaky_relu(a_s[src] + a_d[dst]); segment max over dst
// ---------------------------------------------------------------------------
__global__ void edge_logits(const int* __restrict__ src, const int* __restrict__ dst,
                            const float* __restrict__ as, const float* __restrict__ ad,
                            float* __restrict__ ebuf, float* __restrict__ mmax) {
  int i = blockIdx.x * blockDim.x + threadIdx.x;   // e*HEADS + h
  if (i >= E2 * HEADS) return;
  const int e = i >> 3, h = i & (HEADS - 1);
  int s, d;
  if (e < E_EDGES) { s = src[e]; d = dst[e]; } else { s = d = e - E_EDGES; }
  float v = as[s * HEADS + h] + ad[d * HEADS + h];
  v = (v >= 0.f) ? v : NEG_SLOPE * v;
  ebuf[i] = v;
  atomicMaxF(&mmax[d * HEADS + h], v);
}

// ---------------------------------------------------------------------------
// exp(e - max) and segment sum over dst
// ---------------------------------------------------------------------------
__global__ void edge_exp(const int* __restrict__ dst, float* __restrict__ ebuf,
                         const float* __restrict__ mmax, float* __restrict__ denom) {
  int i = blockIdx.x * blockDim.x + threadIdx.x;
  if (i >= E2 * HEADS) return;
  const int e = i >> 3, h = i & (HEADS - 1);
  const int d = (e < E_EDGES) ? dst[e] : e - E_EDGES;
  float ex = __expf(ebuf[i] - mmax[d * HEADS + h]);
  ebuf[i] = ex;
  atomicAdd(&denom[d * HEADS + h], ex);
}

// ---------------------------------------------------------------------------
// Message aggregation: one wave per edge, 8 channels per lane.
// agg[dst] += alpha * hw[src]
// ---------------------------------------------------------------------------
__global__ void edge_agg(const int* __restrict__ src, const int* __restrict__ dst,
                         const float* __restrict__ hw, const float* __restrict__ ebuf,
                         const float* __restrict__ denom, float* __restrict__ agg) {
  const int e = blockIdx.x * (blockDim.x >> 5) + (threadIdx.x >> 5);
  if (e >= E2) return;
  const int lane = threadIdx.x & 31;
  int s, d;
  if (e < E_EDGES) { s = src[e]; d = dst[e]; } else { s = d = e - E_EDGES; }
  const int ch0 = lane * 8;            // channels [ch0, ch0+8) lie in one head
  const int h   = lane >> 2;           // head = ch0 / CH
  const float alpha = ebuf[e * HEADS + h] / denom[d * HEADS + h];
  const float* __restrict__ hs = hw  + (size_t)s * HID + ch0;
  float*                    od = agg + (size_t)d * HID + ch0;
  __builtin_prefetch(hs, 0, 1);        // global_prefetch_b8
#pragma unroll
  for (int j = 0; j < 8; ++j) atomicAdd(&od[j], alpha * hs[j]);
}

__global__ void bias_relu(float* __restrict__ h, const float* __restrict__ bias) {
  int i = blockIdx.x * blockDim.x + threadIdx.x;
  if (i >= N_NODES * HID) return;
  float v = h[i] + bias[i & (HID - 1)];
  h[i] = (v > 0.f) ? v : 0.f;
}

// ---------------------------------------------------------------------------
// Global mean pool
// ---------------------------------------------------------------------------
__global__ void pool_init(float* __restrict__ gsum, float* __restrict__ gcnt) {
  int i = blockIdx.x * blockDim.x + threadIdx.x;
  if (i < NGRAPH * HID) gsum[i] = 0.f;
  if (i < NGRAPH)       gcnt[i] = 0.f;
}

__global__ void pool_sum(const float* __restrict__ h, const int* __restrict__ batch,
                         float* __restrict__ gsum, float* __restrict__ gcnt) {
  int i = blockIdx.x * blockDim.x + threadIdx.x;
  if (i >= N_NODES * HID) return;
  const int n = i >> 8, d = i & (HID - 1);
  const int g = batch[n];
  atomicAdd(&gsum[g * HID + d], h[i]);
  if (d == 0) atomicAdd(&gcnt[g], 1.0f);
}

__global__ void pool_div(const float* __restrict__ gsum, const float* __restrict__ gcnt,
                         float* __restrict__ gfeat) {
  int i = blockIdx.x * blockDim.x + threadIdx.x;
  if (i >= NGRAPH * HID) return;
  float c = gcnt[i >> 8];
  c = (c > 1.f) ? c : 1.f;
  gfeat[i] = gsum[i] / c;
}

// ---------------------------------------------------------------------------
// LoRA heads: xa[g,r] = g . A[r]   then  out = g W^T + b + SCALING * xa B^T
// ---------------------------------------------------------------------------
__global__ void lora_xa(const float* __restrict__ gfeat, const float* __restrict__ muA,
                        const float* __restrict__ lvA, float* __restrict__ xa) {
  int i = blockIdx.x * blockDim.x + threadIdx.x;   // which*G*R + g*R + r
  if (i >= 2 * NGRAPH * LORA_R) return;
  const int which = i / (NGRAPH * LORA_R);
  const int rem   = i % (NGRAPH * LORA_R);
  const int g = rem / LORA_R, r = rem % LORA_R;
  const float* A = which ? lvA : muA;
  const float* x = gfeat + (size_t)g * HID;
  float acc = 0.f;
#pragma unroll 8
  for (int k = 0; k < HID; ++k) acc += x[k] * A[r * HID + k];
  xa[i] = acc;
}

__global__ void lora_head(const float* __restrict__ gfeat, const float* __restrict__ xa,
                          const float* __restrict__ muW, const float* __restrict__ mub,
                          const float* __restrict__ muB, const float* __restrict__ lvW,
                          const float* __restrict__ lvb, const float* __restrict__ lvB,
                          float* __restrict__ out) {
  int i = blockIdx.x * blockDim.x + threadIdx.x;   // which*G*LAT + g*LAT + j
  if (i >= 2 * NGRAPH * LAT) return;
  const int which = i / (NGRAPH * LAT);
  const int rem   = i % (NGRAPH * LAT);
  const int g = rem / LAT, j = rem % LAT;
  const float* W = which ? lvW : muW;
  const float* b = which ? lvb : mub;
  const float* B = which ? lvB : muB;
  const float* x = gfeat + (size_t)g * HID;
  float acc = b[j];
#pragma unroll 8
  for (int k = 0; k < HID; ++k) acc += x[k] * W[j * HID + k];
  const float* xag = xa + which * NGRAPH * LORA_R + g * LORA_R;
  float lr = 0.f;
#pragma unroll
  for (int r = 0; r < LORA_R; ++r) lr += B[j * LORA_R + r] * xag[r];
  out[i] = acc + SCALING * lr;   // mu first (which=0), then log_var — matches d_out order
}

// ---------------------------------------------------------------------------
// Host launcher
// ---------------------------------------------------------------------------
extern "C" void kernel_launch(void* const* d_in, const int* in_sizes, int n_in,
                              void* d_out, int out_size, void* d_ws, size_t ws_size,
                              hipStream_t stream) {
  (void)in_sizes; (void)n_in; (void)out_size; (void)ws_size;

  const float* x     = (const float*)d_in[0];
  const int*   ei    = (const int*)d_in[1];
  const int*   batch = (const int*)d_in[2];
  const float* enc_W = (const float*)d_in[3];
  const float* enc_b = (const float*)d_in[4];
  const float* gat_W = (const float*)d_in[5];   // [L, HID, HID]
  const float* att_s = (const float*)d_in[6];   // [L, H, C]
  const float* att_d = (const float*)d_in[7];
  const float* gat_b = (const float*)d_in[8];   // [L, HID]
  const float* mu_W  = (const float*)d_in[9];
  const float* mu_b  = (const float*)d_in[10];
  const float* mu_A  = (const float*)d_in[11];
  const float* mu_B  = (const float*)d_in[12];
  const float* lv_W  = (const float*)d_in[13];
  const float* lv_b  = (const float*)d_in[14];
  const float* lv_A  = (const float*)d_in[15];
  const float* lv_B  = (const float*)d_in[16];

  const int* src = ei;
  const int* dst = ei + E_EDGES;

  // workspace layout
  char* ws = (char*)d_ws;
  size_t off = 0;
  auto alloc = [&](size_t bytes) -> char* {
    char* p = ws + off;
    off = (off + bytes + 255) & ~(size_t)255;
    return p;
  };
  float*     hbuf  = (float*)alloc((size_t)N_NODES * HID * 4);   // h (also agg target)
  float*     hw    = (float*)alloc((size_t)N_NODES * HID * 4);   // h @ W^T
  float*     ebuf  = (float*)alloc((size_t)E2 * HEADS * 4);
  float*     as    = (float*)alloc((size_t)N_NODES * HEADS * 4);
  float*     ad    = (float*)alloc((size_t)N_NODES * HEADS * 4);
  float*     mmax  = (float*)alloc((size_t)N_NODES * HEADS * 4);
  float*     denom = (float*)alloc((size_t)N_NODES * HEADS * 4);
  float*     gsum  = (float*)alloc((size_t)NGRAPH * HID * 4);
  float*     gcnt  = (float*)alloc((size_t)NGRAPH * 4);
  float*     gfeat = (float*)alloc((size_t)NGRAPH * HID * 4);
  float*     xa    = (float*)alloc((size_t)2 * NGRAPH * LORA_R * 4);
  _Float16*  encW16 = (_Float16*)alloc((size_t)HID * F_IN * 2);
  _Float16*  gatW16 = (_Float16*)alloc((size_t)NLAYERS * HID * HID * 2);

  const dim3 gblk(32, 8);                 // 8 waves; each wave -> 2 m-tiles
  const dim3 ggrid(N_NODES / 16, 1);      // block covers full 256-wide output slab
  const int  TPB = 256;
  const int  nNH  = N_NODES * HEADS;
  const int  nNHd = N_NODES * HID;
  const int  nEH  = E2 * HEADS;

  // one-time weight conversion to f16 (tiny: ~230K elements)
  cvt_f16<<<(HID * F_IN + TPB - 1) / TPB, TPB, 0, stream>>>(enc_W, encW16, HID * F_IN);
  cvt_f16<<<(NLAYERS * HID * HID + TPB - 1) / TPB, TPB, 0, stream>>>(
      gat_W, gatW16, NLAYERS * HID * HID);

  // node encoder: h = x @ enc_W^T + enc_b
  gemm_wmma<F_IN, true><<<ggrid, gblk, 0, stream>>>(x, encW16, enc_b, hbuf, HID);

  for (int l = 0; l < NLAYERS; ++l) {
    gemm_wmma<HID, false><<<ggrid, gblk, 0, stream>>>(
        hbuf, gatW16 + (size_t)l * HID * HID, nullptr, hw, HID);
    att_coef<<<(nNH + TPB - 1) / TPB, TPB, 0, stream>>>(
        hw, att_s + (size_t)l * HEADS * CH, att_d + (size_t)l * HEADS * CH, as, ad);
    layer_init<<<(nNHd + TPB - 1) / TPB, TPB, 0, stream>>>(hbuf, mmax, denom);
    edge_logits<<<(nEH + TPB - 1) / TPB, TPB, 0, stream>>>(src, dst, as, ad, ebuf, mmax);
    edge_exp<<<(nEH + TPB - 1) / TPB, TPB, 0, stream>>>(dst, ebuf, mmax, denom);
    edge_agg<<<(E2 + 7) / 8, TPB, 0, stream>>>(src, dst, hw, ebuf, denom, hbuf);
    bias_relu<<<(nNHd + TPB - 1) / TPB, TPB, 0, stream>>>(hbuf, gat_b + (size_t)l * HID);
  }

  pool_init<<<(NGRAPH * HID + TPB - 1) / TPB, TPB, 0, stream>>>(gsum, gcnt);
  pool_sum<<<(nNHd + TPB - 1) / TPB, TPB, 0, stream>>>(hbuf, batch, gsum, gcnt);
  pool_div<<<(NGRAPH * HID + TPB - 1) / TPB, TPB, 0, stream>>>(gsum, gcnt, gfeat);
  lora_xa<<<(2 * NGRAPH * LORA_R + TPB - 1) / TPB, TPB, 0, stream>>>(gfeat, mu_A, lv_A, xa);
  lora_head<<<(2 * NGRAPH * LAT + TPB - 1) / TPB, TPB, 0, stream>>>(
      gfeat, xa, mu_W, mu_b, mu_B, lv_W, lv_b, lv_B, (float*)d_out);
}
